// PinderMPNNModel_18425409700022
// MI455X (gfx1250) — matrix operand order repaired and checked by reading, not compiled, not run on hardware
//
#include <hip/hip_runtime.h>
#include <cstdint>
#include <cstddef>

// ============================================================================
// EGNN-style MPNN (PinderMPNN) for gfx1250 / MI455X.
// All dense linear layers run through v_wmma_f32_16x16x32_f16 (f16 in, fp32
// accum); each wave computes a 32x64 output tile (8 WMMA per K-step, B frags
// reused across 2 M-subtiles).  BatchNorm(training) via LDS-atomic partials.
// Scatter-mean via global float atomics.  pos_rec/pos_lig live in d_out.
// ============================================================================

typedef __attribute__((ext_vector_type(16))) _Float16 v16h;
typedef __attribute__((ext_vector_type(8)))  _Float16 v8h;
typedef __attribute__((ext_vector_type(8)))  float    v8f;

#define BN_EPS 1e-5f

// ---------------------------------------------------------------------------
// elementwise / utility kernels
// ---------------------------------------------------------------------------
__global__ void k_fill_zero(float* __restrict__ p, size_t n) {
  size_t i  = (size_t)blockIdx.x * blockDim.x + threadIdx.x;
  size_t st = (size_t)gridDim.x * blockDim.x;
  for (; i < n; i += st) p[i] = 0.0f;
}

__global__ void k_copy_f32(const float* __restrict__ a, float* __restrict__ b, size_t n) {
  size_t i  = (size_t)blockIdx.x * blockDim.x + threadIdx.x;
  size_t st = (size_t)gridDim.x * blockDim.x;
  for (; i < n; i += st) b[i] = a[i];
}

__global__ void k_cast_f16(const float* __restrict__ x, _Float16* __restrict__ y, size_t n) {
  size_t i  = (size_t)blockIdx.x * blockDim.x + threadIdx.x;
  size_t st = (size_t)gridDim.x * blockDim.x;
  for (; i < n; i += st) y[i] = (_Float16)x[i];
}

// h0[n,f] = x[n]*W[f] + b[f]   (din==1), written as f16
__global__ void k_lin_in(const float* __restrict__ x, const float* __restrict__ W,
                         const float* __restrict__ b, _Float16* __restrict__ h, size_t total) {
  size_t i  = (size_t)blockIdx.x * blockDim.x + threadIdx.x;
  size_t st = (size_t)gridDim.x * blockDim.x;
  for (; i < total; i += st) {
    int f = (int)(i & 63);
    size_t n = i >> 6;
    h[i] = (_Float16)(x[n] * W[f] + b[f]);
  }
}

// Weight prep: Bt[n*Kpad + k] = (f16)W[k*N + n], zero-padded for k >= Kreal
__global__ void k_prep_bt(const float* __restrict__ W, _Float16* __restrict__ Bt,
                          int Kreal, int Kpad, int N) {
  size_t total = (size_t)N * Kpad;
  size_t i  = (size_t)blockIdx.x * blockDim.x + threadIdx.x;
  size_t st = (size_t)gridDim.x * blockDim.x;
  for (; i < total; i += st) {
    int n = (int)(i / Kpad);
    int k = (int)(i - (size_t)n * Kpad);
    Bt[i] = (k < Kreal) ? (_Float16)W[(size_t)k * N + n] : (_Float16)0.0f;
  }
}

// ---------------------------------------------------------------------------
// WMMA GEMM: C[M,N](f32) = A[M,K](f16,row-major) * Bt(f16,col-major,[N,K]) + bias
// One wave computes a 32x64 tile: 2 M-subtiles x 4 N-subtiles, so each B
// fragment feeds two v_wmma_f32_16x16x32_f16 ops (12 b128 loads : 8 WMMA).
// A-frag layout (16-bit A 16x32, ISA 7.12.2): lane&15 = row, lane>>4 selects
//   K {0..7,16..23} vs {8..15,24..31}  -> two contiguous 16B loads.
// B-frag layout: lane&15 = col, lane>>4 selects K {0..7,16..23} vs {8..15,24..31}
//   -> two contiguous 16B loads from the pre-transposed weight.
// M need only be a multiple of 16: the second M-subtile clamps its row
// pointer and stores conditionally.
// ---------------------------------------------------------------------------
__device__ __forceinline__ v16h ld16h(const _Float16* p) {
  v16h r;
  ((v8h*)&r)[0] = *(const v8h*)(p);
  ((v8h*)&r)[1] = *(const v8h*)(p + 16);
  return r;
}

#define WMMA_F16(AF, BF, CF) \
  __builtin_amdgcn_wmma_f32_16x16x32_f16(false, (AF), false, (BF), (short)0, (CF), false, false)

__global__ __launch_bounds__(256)
void k_gemm_wmma(const _Float16* __restrict__ A, const _Float16* __restrict__ Bt,
                 const float* __restrict__ bias, float* __restrict__ C,
                 int M, int N, int K) {
  const int lane   = threadIdx.x & 31;
  const int tilesN = N >> 6;                       // 64-wide N tiles
  const int tilesM = (M + 31) >> 5;                // 32-row strips
  const int tile   = blockIdx.x * 8 + (threadIdx.x >> 5);
  const int tm     = tile / tilesN;
  if (tm >= tilesM) return;
  const int tn  = tile - tm * tilesN;
  const int r   = lane & 15;
  const int sel = lane >> 4;

  const int  row0 = tm * 32 + r;
  const bool has1 = (tm * 32 + 16) < M;
  const int  row1 = has1 ? (tm * 32 + 16 + r) : row0;   // clamp on tail strip

  const _Float16* ap0 = A + (size_t)row0 * K + sel * 8;
  const _Float16* ap1 = A + (size_t)row1 * K + sel * 8;
  const int c0 = tn * 64 + r;
  const _Float16* bp0 = Bt + (size_t)(c0     ) * K + sel * 8;
  const _Float16* bp1 = Bt + (size_t)(c0 + 16) * K + sel * 8;
  const _Float16* bp2 = Bt + (size_t)(c0 + 32) * K + sel * 8;
  const _Float16* bp3 = Bt + (size_t)(c0 + 48) * K + sel * 8;

  v8f d00 = {}, d01 = {}, d02 = {}, d03 = {};
  v8f d10 = {}, d11 = {}, d12 = {}, d13 = {};
  for (int k = 0; k < K; k += 32) {
    v16h a0, a1;
    ((v8h*)&a0)[0] = *(const v8h*)(ap0 + k);
    ((v8h*)&a0)[1] = *(const v8h*)(ap0 + k + 16);
    ((v8h*)&a1)[0] = *(const v8h*)(ap1 + k);
    ((v8h*)&a1)[1] = *(const v8h*)(ap1 + k + 16);
    __builtin_prefetch(ap0 + k + 256, 0, 3);   // WGP-scope prefetch on the A streams
    __builtin_prefetch(ap1 + k + 256, 0, 3);
    v16h b0 = ld16h(bp0 + k);
    v16h b1 = ld16h(bp1 + k);
    v16h b2 = ld16h(bp2 + k);
    v16h b3 = ld16h(bp3 + k);
    d00 = WMMA_F16(a0, b0, d00);
    d10 = WMMA_F16(a1, b0, d10);
    d01 = WMMA_F16(a0, b1, d01);
    d11 = WMMA_F16(a1, b1, d11);
    d02 = WMMA_F16(a0, b2, d02);
    d12 = WMMA_F16(a1, b2, d12);
    d03 = WMMA_F16(a0, b3, d03);
    d13 = WMMA_F16(a1, b3, d13);
  }

  // C/D layout: col = lane&15 (per N-subtile), rows = sel*8 + i
  const int rb0 = tm * 32 + sel * 8;
  const int rb1 = tm * 32 + 16 + sel * 8;
  {
    int c = tn * 64 + r;       float bv = bias[c];
    #pragma unroll
    for (int i = 0; i < 8; ++i) C[(size_t)(rb0 + i) * N + c] = d00[i] + bv;
    if (has1) {
      #pragma unroll
      for (int i = 0; i < 8; ++i) C[(size_t)(rb1 + i) * N + c] = d10[i] + bv;
    }
  }
  {
    int c = tn * 64 + 16 + r;  float bv = bias[c];
    #pragma unroll
    for (int i = 0; i < 8; ++i) C[(size_t)(rb0 + i) * N + c] = d01[i] + bv;
    if (has1) {
      #pragma unroll
      for (int i = 0; i < 8; ++i) C[(size_t)(rb1 + i) * N + c] = d11[i] + bv;
    }
  }
  {
    int c = tn * 64 + 32 + r;  float bv = bias[c];
    #pragma unroll
    for (int i = 0; i < 8; ++i) C[(size_t)(rb0 + i) * N + c] = d02[i] + bv;
    if (has1) {
      #pragma unroll
      for (int i = 0; i < 8; ++i) C[(size_t)(rb1 + i) * N + c] = d12[i] + bv;
    }
  }
  {
    int c = tn * 64 + 48 + r;  float bv = bias[c];
    #pragma unroll
    for (int i = 0; i < 8; ++i) C[(size_t)(rb0 + i) * N + c] = d03[i] + bv;
    if (has1) {
      #pragma unroll
      for (int i = 0; i < 8; ++i) C[(size_t)(rb1 + i) * N + c] = d13[i] + bv;
    }
  }
}

// ---------------------------------------------------------------------------
// BatchNorm (training): two-stage column reduction.
// Stage 1: per-block LDS-atomic partials (coalesced element walk), then one
//          global atomicAdd per column per block.
// Stage 2: finalize scale/shift:  scale=g*rsqrt(var+eps), shift=b-mean*scale
// N is always a power of two (64/128/256).
// ---------------------------------------------------------------------------
__global__ void k_bn_partial(const float* __restrict__ X, float* __restrict__ psum,
                             float* __restrict__ psq, size_t total, int N, size_t chunk) {
  extern __shared__ float lds[];              // [N] sums, [N] sumsq
  float* ls = lds;
  float* lq = lds + N;
  for (int c = threadIdx.x; c < 2 * N; c += blockDim.x) lds[c] = 0.0f;
  __syncthreads();
  size_t base = (size_t)blockIdx.x * chunk;   // chunk is a multiple of N
  size_t rem  = total - base;
  size_t cnt  = rem < chunk ? rem : chunk;
  const int mask = N - 1;
  for (size_t i = threadIdx.x; i < cnt; i += blockDim.x) {
    float v = X[base + i];
    int c = (int)(i & (size_t)mask);
    atomicAdd(&ls[c], v);
    atomicAdd(&lq[c], v * v);
  }
  __syncthreads();
  for (int c = threadIdx.x; c < N; c += blockDim.x) {
    atomicAdd(&psum[c], ls[c]);
    atomicAdd(&psq[c],  lq[c]);
  }
}

__global__ void k_bn_finalize(const float* __restrict__ psum, const float* __restrict__ psq,
                              const float* __restrict__ g, const float* __restrict__ b,
                              float* __restrict__ stats, int M, int N) {
  int c = blockIdx.x * blockDim.x + threadIdx.x;
  if (c >= N) return;
  float invM = 1.0f / (float)M;
  float m    = psum[c] * invM;
  float var  = psq[c] * invM - m * m;
  float sc   = g[c] * rsqrtf(var + BN_EPS);
  stats[c]     = sc;
  stats[N + c] = b[c] - m * sc;
}

// y16 = (f16) relu(x*scale + shift)
__global__ void k_bn_relu_f16(const float* __restrict__ X, const float* __restrict__ stats,
                              _Float16* __restrict__ Y, size_t total, int N) {
  size_t i  = (size_t)blockIdx.x * blockDim.x + threadIdx.x;
  size_t st = (size_t)gridDim.x * blockDim.x;
  const int mask = N - 1;
  for (; i < total; i += st) {
    int c = (int)(i & (size_t)mask);
    float v = X[i] * stats[c] + stats[N + c];
    Y[i] = (_Float16)fmaxf(v, 0.0f);
  }
}

// ---------------------------------------------------------------------------
// Edge feature build: feat[e] = [h[dst], h[src], |pos[dst]-pos[src]|, 0-pad]
// ---------------------------------------------------------------------------
__global__ void k_build_edgefeat(const _Float16* __restrict__ h, const float* __restrict__ pos,
                                 const int* __restrict__ src, const int* __restrict__ dst,
                                 _Float16* __restrict__ feat, int E, int d, int Kpad) {
  int e  = blockIdx.x * blockDim.x + threadIdx.x;
  int st = gridDim.x * blockDim.x;
  const int nb = d >> 3;
  for (; e < E; e += st) {
    int sj = src[e], ti = dst[e];
    _Float16* out = feat + (size_t)e * Kpad;
    const v8h* hi = (const v8h*)(h + (size_t)ti * d);
    const v8h* hj = (const v8h*)(h + (size_t)sj * d);
    v8h* o = (v8h*)out;
    for (int i = 0; i < nb; ++i) o[i] = hi[i];
    for (int i = 0; i < nb; ++i) o[nb + i] = hj[i];
    float dx = pos[ti * 3 + 0] - pos[sj * 3 + 0];
    float dy = pos[ti * 3 + 1] - pos[sj * 3 + 1];
    float dz = pos[ti * 3 + 2] - pos[sj * 3 + 2];
    out[2 * d] = (_Float16)sqrtf(dx * dx + dy * dy + dz * dz);
    for (int k = 2 * d + 1; k < Kpad; ++k) out[k] = (_Float16)0.0f;
  }
}

// pos_l2: w[e] = dot(w1[e,:], W[:,0]) + b
__global__ void k_posw(const _Float16* __restrict__ w1, const float* __restrict__ W,
                       const float* __restrict__ b, float* __restrict__ wout, int E, int d) {
  int e  = blockIdx.x * blockDim.x + threadIdx.x;
  int st = gridDim.x * blockDim.x;
  for (; e < E; e += st) {
    const _Float16* row = w1 + (size_t)e * d;
    float s = 0.0f;
    for (int c = 0; c < d; ++c) s += (float)row[c] * W[c];
    wout[e] = s + b[0];
  }
}

__global__ void k_count(const int* __restrict__ dst, float* __restrict__ cnt, int E) {
  int e  = blockIdx.x * blockDim.x + threadIdx.x;
  int st = gridDim.x * blockDim.x;
  for (; e < E; e += st) atomicAdd(&cnt[dst[e]], 1.0f);
}

__global__ void k_msg_agg(const _Float16* __restrict__ msg, const int* __restrict__ dst,
                          float* __restrict__ magg, size_t total, int dshift) {
  size_t i  = (size_t)blockIdx.x * blockDim.x + threadIdx.x;
  size_t st = (size_t)gridDim.x * blockDim.x;
  const size_t mask = ((size_t)1 << dshift) - 1;
  for (; i < total; i += st) {
    size_t e = i >> dshift;
    int    c = (int)(i & mask);
    atomicAdd(&magg[((size_t)dst[e] << dshift) + c], (float)msg[i]);
  }
}

__global__ void k_pos_agg(const float* __restrict__ pos, const int* __restrict__ src,
                          const int* __restrict__ dst, const float* __restrict__ w,
                          float* __restrict__ pagg, int E) {
  int e  = blockIdx.x * blockDim.x + threadIdx.x;
  int st = gridDim.x * blockDim.x;
  for (; e < E; e += st) {
    int sj = src[e], ti = dst[e];
    float we = w[e];
    atomicAdd(&pagg[ti * 3 + 0], (pos[ti * 3 + 0] - pos[sj * 3 + 0]) * we);
    atomicAdd(&pagg[ti * 3 + 1], (pos[ti * 3 + 1] - pos[sj * 3 + 1]) * we);
    atomicAdd(&pagg[ti * 3 + 2], (pos[ti * 3 + 2] - pos[sj * 3 + 2]) * we);
  }
}

// upd_in[n] = [h[n,:], msg_aggr[n,:]/max(cnt,1)] as f16
__global__ void k_build_updin(const _Float16* __restrict__ h, const float* __restrict__ magg,
                              const float* __restrict__ cnt, _Float16* __restrict__ u,
                              size_t total, int dshift) {
  size_t i  = (size_t)blockIdx.x * blockDim.x + threadIdx.x;
  size_t st = (size_t)gridDim.x * blockDim.x;
  const int d = 1 << dshift;
  const size_t mask2 = ((size_t)2 << dshift) - 1;
  for (; i < total; i += st) {
    size_t n = i >> (dshift + 1);
    int    c = (int)(i & mask2);
    if (c < d) {
      u[i] = h[(n << dshift) + c];
    } else {
      float cm = fmaxf(cnt[n], 1.0f);
      u[i] = (_Float16)(magg[(n << dshift) + (c - d)] / cm);
    }
  }
}

__global__ void k_pos_update(float* __restrict__ pos, const float* __restrict__ pagg,
                             const float* __restrict__ cnt, int Nn) {
  int n  = blockIdx.x * blockDim.x + threadIdx.x;
  int st = gridDim.x * blockDim.x;
  for (; n < Nn; n += st) {
    float cm = fmaxf(cnt[n], 1.0f);
    pos[n * 3 + 0] += pagg[n * 3 + 0] / cm;
    pos[n * 3 + 1] += pagg[n * 3 + 1] / cm;
    pos[n * 3 + 2] += pagg[n * 3 + 2] / cm;
  }
}

// ---------------------------------------------------------------------------
// host-side orchestration
// ---------------------------------------------------------------------------
struct Lin { const float* W; const float* b; };
struct Bnp { const float* g; const float* b; };
struct Layer {
  Lin msg_l1; Bnp msg_bn1; Lin msg_l2; Bnp msg_bn2;
  Lin pos_l1; Bnp pos_bn;  Lin pos_l2;
  Lin upd_l1; Bnp upd_bn1; Lin upd_l2; Bnp upd_bn2;
  Lin out;
};
struct WS {
  _Float16 *ef, *ea, *eb, *u16, *na16, *nb16, *h16a, *h16b, *bt;
  float *t0, *n0, *wedge, *magg, *cnt, *pagg, *stats, *psum, *psq;
};

static inline dim3 gsgrid(size_t n, int t = 256) {
  size_t b = (n + (size_t)t - 1) / (size_t)t;
  if (b > 16384) b = 16384;
  if (b == 0) b = 1;
  return dim3((unsigned)b);
}

static void h_fill(float* p, size_t n, hipStream_t s) {
  k_fill_zero<<<gsgrid(n), 256, 0, s>>>(p, n);
}

static void h_gemm(const _Float16* A, const _Float16* Bt, const float* bias, float* C,
                   int M, int N, int K, hipStream_t s) {
  int tiles  = ((M + 31) / 32) * (N / 64);
  int blocks = (tiles + 7) / 8;
  k_gemm_wmma<<<dim3(blocks), 256, 0, s>>>(A, Bt, bias, C, M, N, K);
}

static void h_prep_bt(const float* W, _Float16* Bt, int Kreal, int Kpad, int N, hipStream_t s) {
  k_prep_bt<<<gsgrid((size_t)N * Kpad), 256, 0, s>>>(W, Bt, Kreal, Kpad, N);
}

static void h_bn(const float* X, const Bnp& bp, int M, int N, const WS& w, hipStream_t s) {
  h_fill(w.psum, 1024, s);                        // zero psum[512] + psq[512]
  size_t total = (size_t)M * N;
  size_t chunk = (size_t)256 * N;                 // multiple of N
  int blocks = (int)((total + chunk - 1) / chunk);
  k_bn_partial<<<dim3(blocks), 256, 2 * N * sizeof(float), s>>>(X, w.psum, w.psq, total, N, chunk);
  k_bn_finalize<<<dim3((N + 255) / 256), 256, 0, s>>>(w.psum, w.psq, bp.g, bp.b, w.stats, M, N);
}

static void h_relu16(const float* X, _Float16* Y, size_t total, int N, const WS& w, hipStream_t s) {
  k_bn_relu_f16<<<gsgrid(total), 256, 0, s>>>(X, w.stats, Y, total, N);
}

static inline int ilog2(int v) { int s = 0; while ((1 << s) < v) ++s; return s; }

static void run_layer(const Layer& P, int Nn, int E, const int* src, const int* dst,
                      float* pos, const _Float16* hin, _Float16* hout,
                      int d, int dout, const WS& w, hipStream_t s) {
  const int ds  = ilog2(d);
  const int Kp1 = ((2 * d + 1) + 31) & ~31;

  // ---- message MLP on edges ----
  k_build_edgefeat<<<gsgrid((size_t)E), 256, 0, s>>>(hin, pos, src, dst, w.ef, E, d, Kp1);
  h_prep_bt(P.msg_l1.W, w.bt, 2 * d + 1, Kp1, d, s);
  h_gemm(w.ef, w.bt, P.msg_l1.b, w.t0, E, d, Kp1, s);
  h_bn(w.t0, P.msg_bn1, E, d, w, s);
  h_relu16(w.t0, w.ea, (size_t)E * d, d, w, s);

  h_prep_bt(P.msg_l2.W, w.bt, d, d, d, s);
  h_gemm(w.ea, w.bt, P.msg_l2.b, w.t0, E, d, d, s);
  h_bn(w.t0, P.msg_bn2, E, d, w, s);
  h_relu16(w.t0, w.eb, (size_t)E * d, d, w, s);          // msg (f16)

  // ---- position weight MLP ----
  h_prep_bt(P.pos_l1.W, w.bt, d, d, d, s);
  h_gemm(w.eb, w.bt, P.pos_l1.b, w.t0, E, d, d, s);
  h_bn(w.t0, P.pos_bn, E, d, w, s);
  h_relu16(w.t0, w.ea, (size_t)E * d, d, w, s);          // w1 (f16)
  k_posw<<<gsgrid((size_t)E), 256, 0, s>>>(w.ea, P.pos_l2.W, P.pos_l2.b, w.wedge, E, d);

  // ---- scatter-mean aggregation ----
  h_fill(w.cnt, (size_t)Nn, s);
  h_fill(w.magg, (size_t)Nn * d, s);
  h_fill(w.pagg, (size_t)Nn * 3, s);
  k_count<<<gsgrid((size_t)E), 256, 0, s>>>(dst, w.cnt, E);
  k_msg_agg<<<gsgrid((size_t)E * d), 256, 0, s>>>(w.eb, dst, w.magg, (size_t)E * d, ds);
  k_pos_agg<<<gsgrid((size_t)E), 256, 0, s>>>(pos, src, dst, w.wedge, w.pagg, E);

  // ---- update MLP on nodes ----
  k_build_updin<<<gsgrid((size_t)Nn * 2 * d), 256, 0, s>>>(hin, w.magg, w.cnt, w.u16,
                                                           (size_t)Nn * 2 * d, ds);
  h_prep_bt(P.upd_l1.W, w.bt, 2 * d, 2 * d, d, s);
  h_gemm(w.u16, w.bt, P.upd_l1.b, w.n0, Nn, d, 2 * d, s);
  h_bn(w.n0, P.upd_bn1, Nn, d, w, s);
  h_relu16(w.n0, w.na16, (size_t)Nn * d, d, w, s);

  h_prep_bt(P.upd_l2.W, w.bt, d, d, d, s);
  h_gemm(w.na16, w.bt, P.upd_l2.b, w.n0, Nn, d, d, s);
  h_bn(w.n0, P.upd_bn2, Nn, d, w, s);
  h_relu16(w.n0, w.nb16, (size_t)Nn * d, d, w, s);

  h_prep_bt(P.out.W, w.bt, d, d, dout, s);
  h_gemm(w.nb16, w.bt, P.out.b, w.n0, Nn, dout, d, s);
  k_cast_f16<<<gsgrid((size_t)Nn * dout), 256, 0, s>>>(w.n0, hout, (size_t)Nn * dout);

  // ---- position update ----
  k_pos_update<<<gsgrid((size_t)Nn), 256, 0, s>>>(pos, w.pagg, w.cnt, Nn);
}

static void run_graph(const Lin& lin_in, const Layer* layers, int Nn, int E,
                      const int* src, const int* dst, const float* x, float* pos,
                      const WS& w, hipStream_t s) {
  k_lin_in<<<gsgrid((size_t)Nn * 64), 256, 0, s>>>(x, lin_in.W, lin_in.b, w.h16a, (size_t)Nn * 64);
  _Float16* hin  = w.h16a;
  _Float16* hout = w.h16b;
  const int din[3] = {64, 128, 256};
  const int dou[3] = {128, 256, 512};
  for (int l = 0; l < 3; ++l) {
    run_layer(layers[l], Nn, E, src, dst, pos, hin, hout, din[l], dou[l], w, s);
    _Float16* t = hin; hin = hout; hout = t;
  }
}

// ---------------------------------------------------------------------------
extern "C" void kernel_launch(void* const* d_in, const int* in_sizes, int n_in,
                              void* d_out, int out_size, void* d_ws, size_t ws_size,
                              hipStream_t stream) {
  (void)n_in; (void)out_size; (void)ws_size;

  // ---- unflatten inputs (setup_inputs insertion order, params recursive) ----
  int idx = 0;
  const float* x_rec   = (const float*)d_in[idx++];
  const float* pos_rec = (const float*)d_in[idx++];
  const float* x_lig   = (const float*)d_in[idx++];
  const float* pos_lig = (const float*)d_in[idx++];
  const int*   e_rec   = (const int*)d_in[idx++];
  const int*   e_lig   = (const int*)d_in[idx++];

  auto rdLin = [&](Lin& L) { L.W = (const float*)d_in[idx++]; L.b = (const float*)d_in[idx++]; };
  auto rdBn  = [&](Bnp& B) { B.g = (const float*)d_in[idx++]; B.b = (const float*)d_in[idx++]; };
  Lin lin_rec, lin_lig;
  rdLin(lin_rec);
  rdLin(lin_lig);
  Layer rec[3], lig[3];
  auto rdLayer = [&](Layer& L) {
    rdLin(L.msg_l1); rdBn(L.msg_bn1); rdLin(L.msg_l2); rdBn(L.msg_bn2);
    rdLin(L.pos_l1); rdBn(L.pos_bn);  rdLin(L.pos_l2);
    rdLin(L.upd_l1); rdBn(L.upd_bn1); rdLin(L.upd_l2); rdBn(L.upd_bn2);
    rdLin(L.out);
  };
  for (int l = 0; l < 3; ++l) rdLayer(rec[l]);
  for (int l = 0; l < 3; ++l) rdLayer(lig[l]);

  const int N_rec = in_sizes[0];
  const int N_lig = in_sizes[2];
  const int E_rec = in_sizes[4] / 2;
  const int E_lig = in_sizes[5] / 2;
  const size_t EMAX = (size_t)(E_rec > E_lig ? E_rec : E_lig);
  const size_t NMAX = (size_t)(N_rec > N_lig ? N_rec : N_lig);

  // ---- carve workspace ----
  const size_t DM = 256, KM = 544, DOM = 512;
  char* p = (char*)d_ws;
  auto take = [&](size_t bytes) { char* q = p; p += (bytes + 255) & ~(size_t)255; return q; };
  WS w;
  w.ef    = (_Float16*)take(EMAX * KM * 2);
  w.t0    = (float*)   take(EMAX * DM * 4);
  w.ea    = (_Float16*)take(EMAX * DM * 2);
  w.eb    = (_Float16*)take(EMAX * DM * 2);
  w.u16   = (_Float16*)take(NMAX * 2 * DM * 2);
  w.na16  = (_Float16*)take(NMAX * DM * 2);
  w.nb16  = (_Float16*)take(NMAX * DM * 2);
  w.h16a  = (_Float16*)take(NMAX * DOM * 2);
  w.h16b  = (_Float16*)take(NMAX * DOM * 2);
  w.n0    = (float*)   take(NMAX * DOM * 4);
  w.wedge = (float*)   take(EMAX * 4);
  w.magg  = (float*)   take(NMAX * DM * 4);
  w.cnt   = (float*)   take(NMAX * 4);
  w.pagg  = (float*)   take(NMAX * 3 * 4);
  w.stats = (float*)   take(1024 * 4);
  w.psum  = (float*)   take(1024 * 4);
  w.psq   = w.psum + 512;
  w.bt    = (_Float16*)take(KM * DOM * 2);

  // ---- pos lives in d_out; re-init from inputs each call ----
  float* outp = (float*)d_out;
  float* posR = outp;
  float* posL = outp + (size_t)N_rec * 3;
  k_copy_f32<<<gsgrid((size_t)N_rec * 3), 256, 0, stream>>>(pos_rec, posR, (size_t)N_rec * 3);
  k_copy_f32<<<gsgrid((size_t)N_lig * 3), 256, 0, stream>>>(pos_lig, posL, (size_t)N_lig * 3);

  // ---- run both graphs ----
  run_graph(lin_rec, rec, N_rec, E_rec, e_rec, e_rec + E_rec, x_rec, posR, w, stream);
  run_graph(lin_lig, lig, N_lig, E_lig, e_lig, e_lig + E_lig, x_lig, posL, w, stream);
}
